// FlashBCOP_80178449481802
// MI455X (gfx1250) — matrix operand order repaired and checked by reading, not compiled, not run on hardware
//
#include <hip/hip_runtime.h>
#include <math.h>

// ---------------------------------------------------------------------------
// Types
// ---------------------------------------------------------------------------
typedef __attribute__((ext_vector_type(16))) __bf16        v16bf;
typedef __attribute__((ext_vector_type(8)))  float         v8f;
typedef __attribute__((ext_vector_type(4)))  unsigned int  u32x4;
typedef __attribute__((ext_vector_type(4)))  float         f32x4;

#define IMGS  32
#define C_IN  256
#define C_OUT 256
#define H     64
#define W     64
#define HP    66
#define WP    66

#define BM 64      // spatial tile (one full output row)
#define BN 128     // output-channel tile
#define KC 32      // K chunk (input channels per stage)
#define KPAD 40    // padded LDS row stride in elements (80B -> conflict-free)
#define NSTEPS 72  // 9 taps * 8 chunks

// ---------------------------------------------------------------------------
// bf16 split helpers (round-to-nearest-even)
// ---------------------------------------------------------------------------
__device__ __forceinline__ unsigned short f2bf(float f) {
  unsigned int u = __builtin_bit_cast(unsigned int, f);
  u += 0x7FFFu + ((u >> 16) & 1u);
  return (unsigned short)(u >> 16);
}
__device__ __forceinline__ float bf2f(unsigned short s) {
  unsigned int u = ((unsigned int)s) << 16;
  return __builtin_bit_cast(float, u);
}

__device__ __forceinline__ v8f wmma_bf16(v16bf a, v16bf b, v8f c) {
  // D = A(16x32 bf16) x B(32x16 bf16) + C(16x16 f32)
  return __builtin_amdgcn_wmma_f32_16x16x32_bf16(false, a, false, b,
                                                 (short)0, c, false, false);
}

// ---------------------------------------------------------------------------
// 1) x (NCHW f32) -> wrap-padded NHWC bf16 hi/lo: xp[img][yy][xx][ic]
// ---------------------------------------------------------------------------
__global__ __launch_bounds__(256) void prep_x_kernel(
    const float* __restrict__ x,
    unsigned short* __restrict__ xph,
    unsigned short* __restrict__ xpl) {
  const int yy  = blockIdx.x;   // 0..65
  const int img = blockIdx.y;   // 0..31
  const int ic  = threadIdx.x;  // 0..255
  const int ys  = (yy + H - 1) & (H - 1);   // wrap
  const float* src = x + (((size_t)img * C_IN + ic) * H + ys) * W;
  const size_t dbase = ((size_t)(img * HP + yy) * WP) * C_IN + ic;
  for (int xx = 0; xx < WP; ++xx) {
    const int xs = (xx + W - 1) & (W - 1);
    const float v = src[xs];
    const unsigned short hb = f2bf(v);
    xph[dbase + (size_t)xx * C_IN] = hb;
    xpl[dbase + (size_t)xx * C_IN] = f2bf(v - bf2f(hb));
  }
}

// ---------------------------------------------------------------------------
// 2) Spectral normalization (power iteration, 3 iters) for matrices j=0,1
// ---------------------------------------------------------------------------
__device__ __forceinline__ float block_sum256(float val, float* red) {
  const int tid = threadIdx.x;
  red[tid] = val;
  __syncthreads();
  for (int s = 128; s > 0; s >>= 1) {
    if (tid < s) red[tid] += red[tid + s];
    __syncthreads();
  }
  const float r = red[0];
  __syncthreads();
  return r;
}

__global__ __launch_bounds__(256) void spectral_kernel(
    const float* __restrict__ PQ, float* __restrict__ Wout) {
  __shared__ float u[256];
  __shared__ float v[128];
  __shared__ float red[256];
  const int j = blockIdx.x;  // matrix index 0 or 1
  const float* Wm = PQ + (size_t)j * 256 * 128;
  float* Wo = Wout + (size_t)j * 256 * 128;
  const int tid = threadIdx.x;
  u[tid] = 1.0f / 16.0f;  // 1/sqrt(256)
  __syncthreads();
  for (int it = 0; it < 3; ++it) {
    float sv = 0.0f;
    if (tid < 128) {
      for (int m = 0; m < 256; ++m) sv += Wm[m * 128 + tid] * u[m];
    }
    float nv = block_sum256((tid < 128) ? sv * sv : 0.0f, red);
    nv = sqrtf(nv) + 1e-12f;
    if (tid < 128) v[tid] = sv / nv;
    __syncthreads();
    float su = 0.0f;
    for (int n = 0; n < 128; ++n) su += Wm[tid * 128 + n] * v[n];
    float nu = block_sum256(su * su, red);
    nu = sqrtf(nu) + 1e-12f;
    u[tid] = su / nu;
    __syncthreads();
  }
  float s2 = 0.0f;
  for (int n = 0; n < 128; ++n) s2 += Wm[tid * 128 + n] * v[n];
  const float sigma = block_sum256(u[tid] * s2, red);
  const float inv = 1.0f / (sigma + 1e-12f);
  for (int n = 0; n < 128; ++n) Wo[tid * 128 + n] = Wm[tid * 128 + n] * inv;
}

// ---------------------------------------------------------------------------
// 3) Generic small fp32 GEMM: C = alpha*op(A)op(B) [+ beta*C] [+ gamma*E]
//    op(A)[m,k] = transA ? A[k*M+m] : A[m*K+k]
//    op(B)[k,n] = transB ? B[n*K+k] : B[k*N+n]
//    (all dims multiples of 16 in this pipeline)
// ---------------------------------------------------------------------------
__global__ __launch_bounds__(256) void gemm_f32_kernel(
    float* __restrict__ C, const float* __restrict__ A,
    const float* __restrict__ B, const float* __restrict__ E,
    int M, int N, int K, int transA, int transB,
    float alpha, float beta, float gamma) {
  __shared__ float As[16][17];
  __shared__ float Bs[16][17];
  const int tx = threadIdx.x & 15;
  const int ty = threadIdx.x >> 4;
  const int row = blockIdx.y * 16 + ty;
  const int col = blockIdx.x * 16 + tx;
  float acc = 0.0f;
  for (int k0 = 0; k0 < K; k0 += 16) {
    As[ty][tx] = transA ? A[(size_t)(k0 + tx) * M + row]
                        : A[(size_t)row * K + (k0 + tx)];
    Bs[ty][tx] = transB ? B[(size_t)col * K + (k0 + ty)]
                        : B[(size_t)(k0 + ty) * N + col];
    __syncthreads();
#pragma unroll
    for (int kk = 0; kk < 16; ++kk) acc += As[ty][kk] * Bs[kk][tx];
    __syncthreads();
  }
  const size_t idx = (size_t)row * N + col;
  float r = alpha * acc;
  if (beta != 0.0f) r += beta * C[idx];
  if (E) r += gamma * E[idx];
  C[idx] = r;
}

// ---------------------------------------------------------------------------
// 4) Build S1_0=p0, S1_1=I-p0, S2_0=p1, S2_1=I-p1 (each 256x256, contiguous)
// ---------------------------------------------------------------------------
__global__ __launch_bounds__(256) void make_S_kernel(
    const float* __restrict__ p0, const float* __restrict__ p1,
    float* __restrict__ S) {
  const int i = blockIdx.x * 256 + threadIdx.x;
  const int r = i >> 8, c = i & 255;
  const float eye = (r == c) ? 1.0f : 0.0f;
  const float a = p0[i], b = p1[i];
  S[i] = a;
  S[65536 + i] = eye - a;
  S[2 * 65536 + i] = b;
  S[3 * 65536 + i] = eye - b;
}

// ---------------------------------------------------------------------------
// 5) taps f32 [tap][oc][ic] -> bf16 hi/lo
// ---------------------------------------------------------------------------
__global__ __launch_bounds__(256) void convert_w_kernel(
    const float* __restrict__ taps,
    unsigned short* __restrict__ wh, unsigned short* __restrict__ wl) {
  const int i = blockIdx.x * 256 + threadIdx.x;
  const float v = taps[i];
  const unsigned short hb = f2bf(v);
  wh[i] = hb;
  wl[i] = f2bf(v - bf2f(hb));
}

// ---------------------------------------------------------------------------
// 6) Main conv as implicit GEMM with bf16x3 WMMA (fp32-accurate)
//    M = spatial (one output row of 64), N = oc (128 per block), K = 2304
//    Software pipelined: global loads for chunk s+1 issue before the WMMAs
//    of chunk s, so memory latency hides behind matrix math.
// ---------------------------------------------------------------------------
__global__ __launch_bounds__(256) void conv_wmma_kernel(
    const unsigned short* __restrict__ xph, const unsigned short* __restrict__ xpl,
    const unsigned short* __restrict__ wth, const unsigned short* __restrict__ wtl,
    const float* __restrict__ bias, float* __restrict__ out) {
  __shared__ unsigned short Ah[BM * KPAD];
  __shared__ unsigned short Al[BM * KPAD];
  __shared__ unsigned short Bh[BN * KPAD];
  __shared__ unsigned short Bl[BN * KPAD];

  const int oc_blk = blockIdx.x;  // 0..1
  const int y      = blockIdx.y;  // 0..63
  const int img    = blockIdx.z;  // 0..31

  const int tid  = threadIdx.x;
  const int lane = tid & 31;
  const int wave = tid >> 5;
  const int hh   = lane >> 4;   // lane half (K-half selector)
  const int l16  = lane & 15;
  const int wm   = wave & 1;    // M strip
  const int wn   = wave >> 1;   // N strip

  const int am = tid >> 2, aq = tid & 3;  // A staging: row 0..63, 8-elem chunk
  const int bo = tid >> 1, bq = tid & 1;  // B staging: row 0..127, 16-elem chunk

  // Per-thread fixed parts of the source addresses (32-bit: tensors < 2^31)
  const unsigned a_thread = (unsigned)am * C_IN + (unsigned)aq * 8;
  const unsigned b_thread = (unsigned)(oc_blk * BN + bo) * C_IN + (unsigned)bq * 16;
  const unsigned a_lds = (unsigned)(am * KPAD + aq * 8);
  const unsigned b_lds = (unsigned)(bo * KPAD + bq * 16);

  const v8f vzero = {0.f, 0.f, 0.f, 0.f, 0.f, 0.f, 0.f, 0.f};
  v8f acc[2][2];
#pragma unroll
  for (int i = 0; i < 2; ++i)
#pragma unroll
    for (int j = 0; j < 2; ++j) acc[i][j] = vzero;

  union Frag { u32x4 q[2]; v16bf v; };

  // staged-chunk registers
  u32x4 ra_h, ra_l, rb_h0, rb_h1, rb_l0, rb_l1;

  auto load_chunk = [&](int s) {
    const int tap = s >> 3;         // 0..8
    const int c   = s & 7;          // ic chunk
    const int ky  = tap / 3;
    const int kx  = tap - ky * 3;
    const unsigned asrc =
        (unsigned)((img * HP + (y + ky)) * WP + kx) * C_IN + a_thread +
        (unsigned)c * KC;
    ra_h = *(const u32x4*)&xph[asrc];
    ra_l = *(const u32x4*)&xpl[asrc];
    const unsigned bsrc = (unsigned)tap * (C_OUT * C_IN) + b_thread +
                          (unsigned)c * KC;
    rb_h0 = *(const u32x4*)&wth[bsrc];
    rb_h1 = *(const u32x4*)&wth[bsrc + 8];
    rb_l0 = *(const u32x4*)&wtl[bsrc];
    rb_l1 = *(const u32x4*)&wtl[bsrc + 8];
  };

  load_chunk(0);

  for (int s = 0; s < NSTEPS; ++s) {
    __syncthreads();  // LDS reads of chunk s-1 complete: safe to overwrite
    *(u32x4*)&Ah[a_lds]     = ra_h;
    *(u32x4*)&Al[a_lds]     = ra_l;
    *(u32x4*)&Bh[b_lds]     = rb_h0;
    *(u32x4*)&Bh[b_lds + 8] = rb_h1;
    *(u32x4*)&Bl[b_lds]     = rb_l0;
    *(u32x4*)&Bl[b_lds + 8] = rb_l1;
    __syncthreads();  // chunk s visible in LDS

    // Kick off global loads for chunk s+1; they complete behind the WMMAs.
    load_chunk(s < NSTEPS - 1 ? s + 1 : NSTEPS - 1);

    // --- fragment loads (documented 16-bit WMMA VGPR layouts) ---
    Frag a_hi[2], a_lo[2], b_hi[2], b_lo[2];
#pragma unroll
    for (int i = 0; i < 2; ++i) {
      const int mrow = (wm * 2 + i) * 16 + l16;           // A: lane = M row
      const int base = mrow * KPAD + hh * 8;              // K in [h*8,h*8+8)
      a_hi[i].q[0] = *(const u32x4*)&Ah[base];
      a_hi[i].q[1] = *(const u32x4*)&Ah[base + 16];       // K in [16+h*8,...)
      a_lo[i].q[0] = *(const u32x4*)&Al[base];
      a_lo[i].q[1] = *(const u32x4*)&Al[base + 16];
    }
#pragma unroll
    for (int j = 0; j < 2; ++j) {
      const int ncol = (wn * 2 + j) * 16 + l16;           // B: lane = N col
      const int base = ncol * KPAD + hh * 8;              // K in [h*8,h*8+8) and [16+h*8,...)
      b_hi[j].q[0] = *(const u32x4*)&Bh[base];
      b_hi[j].q[1] = *(const u32x4*)&Bh[base + 16];
      b_lo[j].q[0] = *(const u32x4*)&Bl[base];
      b_lo[j].q[1] = *(const u32x4*)&Bl[base + 16];
    }
    // --- bf16x3 split-precision accumulation ---
#pragma unroll
    for (int i = 0; i < 2; ++i)
#pragma unroll
      for (int j = 0; j < 2; ++j) {
        acc[i][j] = wmma_bf16(a_hi[i].v, b_hi[j].v, acc[i][j]);
        acc[i][j] = wmma_bf16(a_hi[i].v, b_lo[j].v, acc[i][j]);
        acc[i][j] = wmma_bf16(a_lo[i].v, b_hi[j].v, acc[i][j]);
      }
  }

  // --- epilogue: C layout -> per lane 8 consecutive x positions, fixed oc ---
#pragma unroll
  for (int i = 0; i < 2; ++i) {
    const int xbase = (wm * 2 + i) * 16 + hh * 8;
#pragma unroll
    for (int j = 0; j < 2; ++j) {
      const int oc = oc_blk * BN + (wn * 2 + j) * 16 + l16;
      const float bv = bias[oc];
      float* op = out + (unsigned)((img * C_OUT + oc) * H + y) * W + xbase;
      f32x4 lo4 = {acc[i][j][0] + bv, acc[i][j][1] + bv,
                   acc[i][j][2] + bv, acc[i][j][3] + bv};
      f32x4 hi4 = {acc[i][j][4] + bv, acc[i][j][5] + bv,
                   acc[i][j][6] + bv, acc[i][j][7] + bv};
      *(f32x4*)op = lo4;
      *(f32x4*)(op + 4) = hi4;
    }
  }
}

// ---------------------------------------------------------------------------
// Host orchestration
// ---------------------------------------------------------------------------
extern "C" void kernel_launch(void* const* d_in, const int* in_sizes, int n_in,
                              void* d_out, int out_size, void* d_ws, size_t ws_size,
                              hipStream_t stream) {
  (void)in_sizes; (void)n_in; (void)out_size; (void)ws_size;
  const float* x    = (const float*)d_in[0];
  const float* PQ   = (const float*)d_in[1];
  const float* bias = (const float*)d_in[2];
  float* out = (float*)d_out;

  char* ws = (char*)d_ws;
  size_t off = 0;
  auto alloc = [&](size_t bytes) -> void* {
    off = (off + 255) & ~(size_t)255;
    void* p = ws + off;
    off += bytes;
    return p;
  };

  unsigned short* xph = (unsigned short*)alloc((size_t)IMGS * HP * WP * C_IN * 2);
  unsigned short* xpl = (unsigned short*)alloc((size_t)IMGS * HP * WP * C_IN * 2);
  float* Wn   = (float*)alloc((size_t)2 * 256 * 128 * 4);
  float* Wp   = (float*)alloc((size_t)2 * 256 * 128 * 4);
  float* T    = (float*)alloc((size_t)128 * 128 * 4);
  float* proj = (float*)alloc((size_t)2 * 65536 * 4);
  float* S    = (float*)alloc((size_t)4 * 65536 * 4);
  float* Ab   = (float*)alloc((size_t)4 * 65536 * 4);
  float* taps = (float*)alloc((size_t)9 * 65536 * 4);
  unsigned short* wth = (unsigned short*)alloc((size_t)9 * 65536 * 2);
  unsigned short* wtl = (unsigned short*)alloc((size_t)9 * 65536 * 2);

  auto gemm = [&](float* Cp, const float* Ap, const float* Bp, const float* Ep,
                  int M, int N, int K, int tA, int tB,
                  float alpha, float beta, float gamma) {
    dim3 grid(N / 16, M / 16);
    gemm_f32_kernel<<<grid, 256, 0, stream>>>(Cp, Ap, Bp, Ep, M, N, K, tA, tB,
                                              alpha, beta, gamma);
  };

  // 1) pad + split input to NHWC bf16 hi/lo
  prep_x_kernel<<<dim3(HP, IMGS), 256, 0, stream>>>(x, xph, xpl);

  // 2) spectral normalization (only matrices 0 and 1 affect the output)
  spectral_kernel<<<2, 256, 0, stream>>>(PQ, Wn);

  // 3) Bjorck: W = 1.5 W - 0.5 W (W^T W), 10 iterations, ping-pong
  const float* Wfin[2];
  for (int j = 0; j < 2; ++j) {
    float* Wa = Wn + (size_t)j * 256 * 128;
    float* Wb = Wp + (size_t)j * 256 * 128;
    for (int it = 0; it < 10; ++it) {
      gemm(T, Wa, Wa, nullptr, 128, 128, 256, 1, 0, 1.0f, 0.0f, 0.0f);   // T = W^T W
      gemm(Wb, Wa, T, Wa, 256, 128, 128, 0, 0, -0.5f, 0.0f, 1.5f);       // W' = -0.5 W T + 1.5 W
      float* tmp = Wa; Wa = Wb; Wb = tmp;
    }
    Wfin[j] = Wa;
  }

  // 4) proj_j = W_j W_j^T (256x256)
  gemm(proj,         Wfin[0], Wfin[0], nullptr, 256, 256, 128, 0, 1, 1.0f, 0.0f, 0.0f);
  gemm(proj + 65536, Wfin[1], Wfin[1], nullptr, 256, 256, 128, 0, 1, 1.0f, 0.0f, 0.0f);

  // 5) S1_b, S2_c and A_{b,c} = S1_b @ S2_c
  make_S_kernel<<<256, 256, 0, stream>>>(proj, proj + 65536, S);
  for (int b = 0; b < 2; ++b)
    for (int c = 0; c < 2; ++c)
      gemm(Ab + (size_t)(b * 2 + c) * 65536, S + (size_t)b * 65536,
           S + (size_t)(2 + c) * 65536, nullptr, 256, 256, 256, 0, 0,
           1.0f, 0.0f, 0.0f);

  // 6) taps(y,x) = sum_{y1+y2=y, x1+x2=x} A_{y2,x2} @ A_{y1,x1}
  for (int yk = 0; yk < 3; ++yk)
    for (int xk = 0; xk < 3; ++xk) {
      float* tp = taps + (size_t)(yk * 3 + xk) * 65536;
      bool first = true;
      for (int y1 = 0; y1 <= 1; ++y1) {
        const int y2 = yk - y1;
        if (y2 < 0 || y2 > 1) continue;
        for (int x1 = 0; x1 <= 1; ++x1) {
          const int x2 = xk - x1;
          if (x2 < 0 || x2 > 1) continue;
          gemm(tp, Ab + (size_t)(y2 * 2 + x2) * 65536,
               Ab + (size_t)(y1 * 2 + x1) * 65536, nullptr,
               256, 256, 256, 0, 0, 1.0f, first ? 0.0f : 1.0f, 0.0f);
          first = false;
        }
      }
    }

  // 7) weight -> bf16 hi/lo [tap][oc][ic]
  convert_w_kernel<<<(9 * 65536) / 256, 256, 0, stream>>>(taps, wth, wtl);

  // 8) main conv: implicit GEMM with bf16x3 WMMA, software pipelined
  conv_wmma_kernel<<<dim3(2, H, IMGS), 256, 0, stream>>>(xph, xpl, wth, wtl,
                                                         bias, out);
}
